// TinySwinTransformerBlock_19121194402484
// MI455X (gfx1250) — matrix operand order repaired and checked by reading, not compiled, not run on hardware
//
#include <hip/hip_runtime.h>

// ---------------------------------------------------------------------------
// Types / helpers
// ---------------------------------------------------------------------------
typedef __attribute__((ext_vector_type(16))) __bf16 bf16x16;
typedef __attribute__((ext_vector_type(8)))  float  f32x8;

union ABFrag { bf16x16 v; unsigned u[8]; };

#define WMMA_BF16(A, B, C) \
  __builtin_amdgcn_wmma_f32_16x16x32_bf16(false, (A), false, (B), (short)0, (C), false, false)

__device__ __forceinline__ __bf16 f2bf(float f) {
  unsigned u = __builtin_bit_cast(unsigned, f);
  u += 0x7FFFu + ((u >> 16) & 1u);                 // round-to-nearest-even
  unsigned short s = (unsigned short)(u >> 16);
  return __builtin_bit_cast(__bf16, s);
}

__device__ __forceinline__ f32x8 zero8() {
  f32x8 z;
#pragma unroll
  for (int i = 0; i < 8; ++i) z[i] = 0.0f;
  return z;
}

// Problem constants
// B=64, H=W=56, C=128, WS=7, SS=3, HEADS=4, HEAD_DIM=32, N=49, HID=512
#define MROWS 200704          // B*H*W == B*nW*49
#define SCALE_Q 0.17677669529663689f
// padded qkv: [4096 windows][64 rows][384 cols] bf16
#define QKV_WSTRIDE (64 * 384)

// ---------------------------------------------------------------------------
// fp32 -> bf16 weight conversion
// ---------------------------------------------------------------------------
__global__ __launch_bounds__(256) void cvt_kernel(const float* __restrict__ in,
                                                  __bf16* __restrict__ out, int n) {
  int i = blockIdx.x * 256 + threadIdx.x;
  if (i < n) out[i] = f2bf(in[i]);
}

// ---------------------------------------------------------------------------
// Zero pad rows 49..63 of each window's [64,384] qkv tile (8B chunks).
// chunks per window = 15*384*2/8 = 1440 ; total = 4096*1440 = 5,898,240
// ---------------------------------------------------------------------------
__global__ __launch_bounds__(256) void padzero_kernel(unsigned long long* __restrict__ q) {
  size_t c = (size_t)blockIdx.x * 256 + threadIdx.x;
  size_t win = c / 1440, within = c % 1440;
  // ull index: (win*64*384 + 49*384) elems / 4 elems-per-ull
  q[win * 6144 + 4704 + within] = 0ull;
}

// ---------------------------------------------------------------------------
// LayerNorm (+ optional shift/window-partition gather).  1 wave per row.
// MODE 0: output row r is windowed index; source pixel = roll(-3,-3) gather.
// MODE 1: identity rows (LN2 on residual stream).
// ---------------------------------------------------------------------------
template <int MODE>
__global__ __launch_bounds__(256) void ln_kernel(const float* __restrict__ x,
                                                 const float* __restrict__ g,
                                                 const float* __restrict__ bt,
                                                 __bf16* __restrict__ out) {
  const int lane = threadIdx.x & 31, wv = threadIdx.x >> 5;
  const int r = blockIdx.x * 8 + wv;
  size_t src;
  if (MODE == 0) {
    int win = r / 49, n = r % 49;
    int b = win >> 6, wi = win & 63, wh = wi >> 3, ww = wi & 7;
    int ii = n / 7, jj = n % 7;
    int sh = (wh * 7 + ii + 3) % 56;          // roll(-3): read (h+3)%56
    int sw = (ww * 7 + jj + 3) % 56;
    src = ((size_t)b * 3136 + sh * 56 + sw) * 128;
  } else {
    src = (size_t)r * 128;
  }
  float4 v = ((const float4*)(x + src))[lane];         // 4 channels / lane
  float s = v.x + v.y + v.z + v.w;
  float q = v.x * v.x + v.y * v.y + v.z * v.z + v.w * v.w;
#pragma unroll
  for (int off = 16; off > 0; off >>= 1) {
    s += __shfl_xor(s, off, 32);
    q += __shfl_xor(q, off, 32);
  }
  float mu = s * 0.0078125f;
  float var = q * 0.0078125f - mu * mu;
  float rs = rsqrtf(var + 1e-5f);
  float4 gg = ((const float4*)g)[lane];
  float4 bb = ((const float4*)bt)[lane];
  __bf16* o = out + (size_t)r * 128 + lane * 4;
  o[0] = f2bf((v.x - mu) * rs * gg.x + bb.x);
  o[1] = f2bf((v.y - mu) * rs * gg.y + bb.y);
  o[2] = f2bf((v.z - mu) * rs * gg.z + bb.z);
  o[3] = f2bf((v.w - mu) * rs * gg.w + bb.w);
}

// ---------------------------------------------------------------------------
// Tiled WMMA GEMM:  out = epilogue(A[M,K] @ W[K,N] + bias)
// Block tile 128x128, 8 waves (4Mx2N), each wave 32x64 = 2x4 WMMA tiles.
// MODE 0: QKV     (bias, q-cols * SCALE, bf16 out, PADDED window layout)
// MODE 1: proj    (bias, window-reverse + roll(+3) scatter, += resid -> fp32)
// MODE 2: MLP1    (bias, exact-erf GELU, bf16 out)
// MODE 3: MLP2    (bias, in-place fp32 residual add on outf)
// ---------------------------------------------------------------------------
template <int MODE>
__global__ __launch_bounds__(256) void gemm_wmma(const __bf16* __restrict__ A,
                                                 const __bf16* __restrict__ W,
                                                 const float* __restrict__ bias,
                                                 __bf16* __restrict__ outb,
                                                 float* __restrict__ outf,
                                                 const float* __restrict__ resid,
                                                 int M, int N, int K) {
  constexpr int BM = 128, BN = 128, BK = 32, AS = 40, BSS = 40;
  __shared__ __bf16 As[BM * AS];        // [row][k] padded
  __shared__ __bf16 Bs[BN * BSS];       // transposed: [n][k] padded
  const int tid = threadIdx.x;
  const int lane = tid & 31, wv = tid >> 5;
  const int hl = lane >> 4, lr = lane & 15;
  const int wm = (wv >> 1) * 32;        // wave M offset (4 waves)
  const int wn = (wv & 1) * 64;         // wave N offset (2 waves)
  const int bm = blockIdx.y * BM, bn = blockIdx.x * BN;

  f32x8 acc[2][4];
#pragma unroll
  for (int mt = 0; mt < 2; ++mt)
#pragma unroll
    for (int nt = 0; nt < 4; ++nt) acc[mt][nt] = zero8();

  for (int k0 = 0; k0 < K; k0 += BK) {
    // stage A tile 128x32 (b64 chunks)
    for (int ch = tid; ch < BM * BK / 4; ch += 256) {
      int row = ch >> 3, kc = (ch & 7) << 2;
      *(unsigned long long*)&As[row * AS + kc] =
          *(const unsigned long long*)(A + (size_t)(bm + row) * K + k0 + kc);
    }
    // stage B tile 32x128, transposed into Bs[n][k]
    for (int ch = tid; ch < BK * BN / 4; ch += 256) {
      int kk = ch >> 5, n4 = (ch & 31) << 2;
      unsigned long long d =
          *(const unsigned long long*)(W + (size_t)(k0 + kk) * N + bn + n4);
      __bf16 tmp[4];
      *(unsigned long long*)tmp = d;
      Bs[(n4 + 0) * BSS + kk] = tmp[0];
      Bs[(n4 + 1) * BSS + kk] = tmp[1];
      Bs[(n4 + 2) * BSS + kk] = tmp[2];
      Bs[(n4 + 3) * BSS + kk] = tmp[3];
    }
    if (k0 + BK < K)  // hint next A k-slice into cache (global_prefetch_b8)
      __builtin_prefetch(A + (size_t)(bm + (tid & 127)) * K + k0 + BK, 0, 1);
    __syncthreads();

    ABFrag afr[2], bfr[4];
#pragma unroll
    for (int mt = 0; mt < 2; ++mt)
#pragma unroll
      for (int p = 0; p < 8; ++p) {
        int kb = ((p >> 2) << 4) + (hl << 3) + ((p & 3) << 1);
        afr[mt].u[p] = *(const unsigned*)&As[(wm + mt * 16 + lr) * AS + kb];
      }
#pragma unroll
    for (int nt = 0; nt < 4; ++nt)
#pragma unroll
      for (int p = 0; p < 8; ++p) {
        int kb = ((p >> 2) << 4) + (hl << 3) + ((p & 3) << 1);
        bfr[nt].u[p] = *(const unsigned*)&Bs[(wn + nt * 16 + lr) * BSS + kb];
      }
#pragma unroll
    for (int mt = 0; mt < 2; ++mt)
#pragma unroll
      for (int nt = 0; nt < 4; ++nt)
        acc[mt][nt] = WMMA_BF16(afr[mt].v, bfr[nt].v, acc[mt][nt]);
    __syncthreads();
  }

  // epilogue
#pragma unroll
  for (int mt = 0; mt < 2; ++mt)
#pragma unroll
    for (int i = 0; i < 8; ++i) {
      int row = bm + wm + mt * 16 + hl * 8 + i;
      size_t obase = 0;
      if (MODE == 0) {  // padded per-window layout [win][64][384]
        int win = row / 49, n = row % 49;
        obase = (size_t)win * QKV_WSTRIDE + (size_t)n * 384;
      }
      if (MODE == 1) {  // window-reverse + roll(+3,+3)
        int win = row / 49, n = row % 49;
        int b = win >> 6, wi = win & 63, wh = wi >> 3, ww = wi & 7;
        int ii = n / 7, jj = n % 7;
        int hh = (wh * 7 + ii + 3) % 56;
        int w2 = (ww * 7 + jj + 3) % 56;
        obase = ((size_t)b * 3136 + hh * 56 + w2) * 128;
      }
#pragma unroll
      for (int nt = 0; nt < 4; ++nt) {
        int col = bn + wn + nt * 16 + lr;
        float v = acc[mt][nt][i] + bias[col];
        if (MODE == 0) {
          if (col < 128) v *= SCALE_Q;          // q * head_dim^-0.5
          outb[obase + col] = f2bf(v);
        } else if (MODE == 1) {
          outf[obase + col] = resid[obase + col] + v;
        } else if (MODE == 2) {
          v = 0.5f * v * (1.0f + erff(v * 0.70710678118654752f));
          outb[(size_t)row * N + col] = f2bf(v);
        } else {
          size_t idx = (size_t)row * 128 + col;
          outf[idx] = outf[idx] + v;
        }
      }
    }
}

// ---------------------------------------------------------------------------
// Windowed attention: one (window, head-pair) per 64-thread block.
// qkv is PADDED [win][64][384] with rows 49..63 zeroed -> all fragment loads
// are unconditional contiguous dwords (compiler can fuse into b128 clauses).
// Each wave handles one head: S = Q K^T (16 WMMA), masked softmax, O = P V
// (16 WMMA).  Per row: [q(0:128)|k(128:256)|v(256:384)], col = h*32+d.
// ---------------------------------------------------------------------------
__global__ __launch_bounds__(64) void attn_kernel(const __bf16* __restrict__ qkv,
                                                  __bf16* __restrict__ out) {
  __shared__ float  Sb[2][49 * 64];
  __shared__ __bf16 Pb[2][64 * 64];
  __shared__ __bf16 Vt[2][32 * 64];   // [d][k] transposed V
  __shared__ int    cnt[49];
  const int tid = threadIdx.x, lane = tid & 31, hloc = tid >> 5;
  const int h = blockIdx.y * 2 + hloc;
  const int win = blockIdx.x;
  const int wi = win & 63, wh = wi >> 3, ww = wi & 7;
  const int hl = lane >> 4, lr = lane & 15;
  const __bf16* base = qkv + (size_t)win * QKV_WSTRIDE;
  float*  Sh = Sb[hloc];
  __bf16* Ph = Pb[hloc];
  __bf16* Vh = Vt[hloc];

  if (tid < 49) {  // Swin shift-mask region id for pixel n of this window
    int ii = tid / 7, jj = tid % 7;
    int gr = wh * 7 + ii, gc = ww * 7 + jj;
    int rr = (gr < 49) ? 0 : ((gr < 53) ? 1 : 2);
    int rc = (gc < 49) ? 0 : ((gc < 53) ? 1 : 2);
    cnt[tid] = rr * 3 + rc;
  }
  // stage V transposed: Vh[d*64 + k]  (pad rows are zero in qkv)
#pragma unroll 4
  for (int r = 0; r < 64; ++r)
    Vh[lane * 64 + r] = base[(size_t)r * 384 + 256 + h * 32 + lane];
  __syncthreads();

  // ---- S = Q K^T (head_dim=32 => one k-step per tile) ----
#pragma unroll
  for (int mt = 0; mt < 4; ++mt) {
    ABFrag afr;
    int qr = mt * 16 + lr;
#pragma unroll
    for (int p = 0; p < 8; ++p) {
      int d = ((p >> 2) << 4) + (hl << 3) + ((p & 3) << 1);
      afr.u[p] = *(const unsigned*)(base + (size_t)qr * 384 + h * 32 + d);
    }
#pragma unroll
    for (int nt = 0; nt < 4; ++nt) {
      ABFrag bfr;
      int kr = nt * 16 + lr;
#pragma unroll
      for (int p = 0; p < 8; ++p) {
        int d = ((p >> 2) << 4) + (hl << 3) + ((p & 3) << 1);
        bfr.u[p] = *(const unsigned*)(base + (size_t)kr * 384 + 128 + h * 32 + d);
      }
      f32x8 c = zero8();
      c = WMMA_BF16(afr.v, bfr.v, c);
#pragma unroll
      for (int i = 0; i < 8; ++i) {
        int r = mt * 16 + hl * 8 + i;
        if (r < 49) Sh[r * 64 + nt * 16 + lr] = c[i];
      }
    }
  }
  __syncthreads();

  // ---- masked softmax, P = softmax(S + mask) in bf16 ----
  for (int r = lane; r < 49; r += 32) {
    int cn = cnt[r];
    float mx = -1e30f;
    for (int m = 0; m < 49; ++m) {
      float s = Sh[r * 64 + m] + ((cnt[m] != cn) ? -100.0f : 0.0f);
      mx = fmaxf(mx, s);
    }
    float sum = 0.0f;
    for (int m = 0; m < 49; ++m) {
      float e = __expf(Sh[r * 64 + m] + ((cnt[m] != cn) ? -100.0f : 0.0f) - mx);
      sum += e;
      Sh[r * 64 + m] = e;
    }
    float inv = 1.0f / sum;
    for (int m = 0; m < 49; ++m) Ph[r * 64 + m] = f2bf(Sh[r * 64 + m] * inv);
    for (int m = 49; m < 64; ++m) Ph[r * 64 + m] = f2bf(0.0f);
  }
  for (int r = 49 + lane; r < 64; r += 32)  // keep pad rows finite
    for (int m = 0; m < 64; ++m) Ph[r * 64 + m] = f2bf(0.0f);
  __syncthreads();

  // ---- O = P V  (K=64 => 2 k-steps) ----
#pragma unroll
  for (int mt = 0; mt < 4; ++mt) {
    f32x8 oa[2];
    oa[0] = zero8();
    oa[1] = zero8();
#pragma unroll
    for (int kt = 0; kt < 2; ++kt) {
      ABFrag afr;
#pragma unroll
      for (int p = 0; p < 8; ++p) {
        int kk = kt * 32 + ((p >> 2) << 4) + (hl << 3) + ((p & 3) << 1);
        afr.u[p] = *(const unsigned*)&Ph[(mt * 16 + lr) * 64 + kk];
      }
#pragma unroll
      for (int nt = 0; nt < 2; ++nt) {
        ABFrag bfr;
#pragma unroll
        for (int p = 0; p < 8; ++p) {
          int kk = kt * 32 + ((p >> 2) << 4) + (hl << 3) + ((p & 3) << 1);
          bfr.u[p] = *(const unsigned*)&Vh[(nt * 16 + lr) * 64 + kk];
        }
        oa[nt] = WMMA_BF16(afr.v, bfr.v, oa[nt]);
      }
    }
#pragma unroll
    for (int nt = 0; nt < 2; ++nt)
#pragma unroll
      for (int i = 0; i < 8; ++i) {
        int r = mt * 16 + hl * 8 + i;
        if (r < 49)
          out[((size_t)win * 49 + r) * 128 + h * 32 + nt * 16 + lr] =
              f2bf(oa[nt][i]);
      }
  }
}

// ---------------------------------------------------------------------------
// Host-side orchestration
// ---------------------------------------------------------------------------
extern "C" void kernel_launch(void* const* d_in, const int* in_sizes, int n_in,
                              void* d_out, int out_size, void* d_ws, size_t ws_size,
                              hipStream_t stream) {
  const float* x      = (const float*)d_in[0];
  const float* g1     = (const float*)d_in[1];
  const float* bt1    = (const float*)d_in[2];
  const float* w_qkv  = (const float*)d_in[3];
  const float* b_qkv  = (const float*)d_in[4];
  const float* w_proj = (const float*)d_in[5];
  const float* b_proj = (const float*)d_in[6];
  const float* g2     = (const float*)d_in[7];
  const float* bt2    = (const float*)d_in[8];
  const float* w1     = (const float*)d_in[9];
  const float* b1     = (const float*)d_in[10];
  const float* w2     = (const float*)d_in[11];
  const float* b2     = (const float*)d_in[12];
  float* out = (float*)d_out;

  // workspace layout (bytes)
  const size_t SZ1 = (size_t)MROWS * 128 * 2;  // bf16 [M,128] (xw / LN2 out)
  const size_t SZ2 = (size_t)MROWS * 512 * 2;  // bf16: qkv padded (201MB) / h1 (205MB)
  const size_t SZ3 = (size_t)MROWS * 128 * 2;  // bf16 [M,128] (attn out)
  char* ws = (char*)d_ws;
  __bf16* buf1 = (__bf16*)ws;
  __bf16* buf2 = (__bf16*)(ws + SZ1);
  __bf16* buf3 = (__bf16*)(ws + SZ1 + SZ2);
  __bf16* wqkv_b  = (__bf16*)(ws + SZ1 + SZ2 + SZ3);
  __bf16* wproj_b = wqkv_b + 128 * 384;
  __bf16* w1_b    = wproj_b + 128 * 128;
  __bf16* w2_b    = w1_b + 128 * 512;
  (void)in_sizes; (void)n_in; (void)out_size; (void)ws_size;

  // 1) weights fp32 -> bf16
  cvt_kernel<<<(128 * 384 + 255) / 256, 256, 0, stream>>>(w_qkv, wqkv_b, 128 * 384);
  cvt_kernel<<<(128 * 128 + 255) / 256, 256, 0, stream>>>(w_proj, wproj_b, 128 * 128);
  cvt_kernel<<<(128 * 512 + 255) / 256, 256, 0, stream>>>(w1, w1_b, 128 * 512);
  cvt_kernel<<<(512 * 128 + 255) / 256, 256, 0, stream>>>(w2, w2_b, 512 * 128);

  // 2) LN1 + shift + window partition -> buf1 (bf16)
  ln_kernel<0><<<MROWS / 8, 256, 0, stream>>>(x, g1, bt1, buf1);

  // 3) zero pad rows of qkv, then QKV GEMM -> buf2 (padded [4096][64][384])
  padzero_kernel<<<23040, 256, 0, stream>>>((unsigned long long*)buf2);
  gemm_wmma<0><<<dim3(3, MROWS / 128), 256, 0, stream>>>(
      buf1, wqkv_b, b_qkv, buf2, nullptr, nullptr, MROWS, 384, 128);

  // 4) windowed attention -> buf3 (bf16 [M,128])
  attn_kernel<<<dim3(4096, 2), 64, 0, stream>>>(buf2, buf3);

  // 5) proj GEMM + window reverse + roll + residual -> d_out (fp32 x')
  gemm_wmma<1><<<dim3(1, MROWS / 128), 256, 0, stream>>>(
      buf3, wproj_b, b_proj, nullptr, out, x, MROWS, 128, 128);

  // 6) LN2 on x' -> buf1 (bf16)
  ln_kernel<1><<<MROWS / 8, 256, 0, stream>>>(out, g2, bt2, buf1);

  // 7) MLP1 GEMM + GELU: [M,128]@[128,512] -> buf2 (bf16)
  gemm_wmma<2><<<dim3(4, MROWS / 128), 256, 0, stream>>>(
      buf1, w1_b, b1, buf2, nullptr, nullptr, MROWS, 512, 128);

  // 8) MLP2 GEMM + residual in-place on d_out: [M,512]@[512,128]
  gemm_wmma<3><<<dim3(1, MROWS / 128), 256, 0, stream>>>(
      buf2, w2_b, b2, nullptr, out, nullptr, MROWS, 128, 512);
}
